// DeepDDT_29033978921242
// MI455X (gfx1250) — compile-verified, matrix operand order checked by reading, not run on hardware
//
#include <hip/hip_runtime.h>
#include <hip/hip_bf16.h>

// ---------------------------------------------------------------------------
// DeepDDT forward, fused/pruned for MI455X (gfx1250, wave32, WMMA).
// GEMMs: bf16 v_wmma_f32_16x16x32_bf16, fp32 accumulate, 128x128 block tile,
// A operand streamed into LDS with GLOBAL_LOAD_ASYNC_TO_LDS_B128 (ASYNCcnt).
// Nodes 7..14 calc GEMMs are skipped (outputs provably unused).
// ---------------------------------------------------------------------------

typedef __bf16 bf16;
typedef __attribute__((ext_vector_type(8)))  __bf16 v8bf;
typedef __attribute__((ext_vector_type(16))) __bf16 v16bf;
typedef __attribute__((ext_vector_type(8)))  float  v8f;

#define BATCH 8192
#define DIM   512
#define OUTD  64
#define NLEAF 16

#define MT 128
#define NT 128
#define KT 32

// ---------------------------------------------------------------------------
// fp32 -> bf16 conversion
// ---------------------------------------------------------------------------
__global__ __launch_bounds__(256)
void k_f32_to_bf16(const float* __restrict__ in, bf16* __restrict__ out, unsigned n) {
    unsigned i = blockIdx.x * 256u + threadIdx.x;
    if (i < n) out[i] = (bf16)in[i];
}

// ---------------------------------------------------------------------------
// column mean over M rows of a bf16 [M, N] matrix (one block per column)
// ---------------------------------------------------------------------------
__global__ __launch_bounds__(256)
void k_colmean_bf16(const bf16* __restrict__ A, int M, int N, float* __restrict__ mean) {
    int col = blockIdx.x;
    float s = 0.f;
    for (int r = threadIdx.x; r < M; r += 256)
        s += (float)A[(size_t)r * N + col];
    __shared__ float red[256];
    red[threadIdx.x] = s;
    __syncthreads();
    for (int w = 128; w > 0; w >>= 1) {
        if ((int)threadIdx.x < w) red[threadIdx.x] += red[threadIdx.x + w];
        __syncthreads();
    }
    if (threadIdx.x == 0) mean[col] = red[0] / (float)M;
}

// ---------------------------------------------------------------------------
// WMMA GEMM:  Y[M,N] = act( concat(A0[:,0:K0], A1[:,0:K1]) @ W[N,K]^T + bias )
//   A0/A1 bf16 row-major (K0,K1 multiples of 32); W fp32 row-major [N, K0+K1]
//   128x128 tile per 256-thread block; 8 waves, each wave owns a 32x64 slab
//   (2 A-frags x 4 B-frags -> 8 WMMA per 32-deep K-step).
//   A tile copied global->LDS with async b128 transfers (raw bf16 bytes);
//   B tile goes through VGPRs for fp32->bf16 conversion.
// ---------------------------------------------------------------------------
__global__ __launch_bounds__(256)
void k_gemm_bf16(const bf16* __restrict__ A0, int K0,
                 const bf16* __restrict__ A1, int K1,
                 const float* __restrict__ W,
                 const float* __restrict__ bias,
                 float* __restrict__ outF,
                 bf16*  __restrict__ outH,
                 int N, int relu)
{
    const int K = K0 + K1;
    __shared__ __align__(16) bf16 As[MT][KT];
    __shared__ __align__(16) bf16 Bs[NT][KT];

    const int tid   = threadIdx.x;
    const int wave  = tid >> 5;
    const int lane  = tid & 31;
    const int tileM = blockIdx.x * MT;
    const int tileN = blockIdx.y * NT;

    const int rowblk = (wave & 3) * 32;   // wave's 32-row slab
    const int colblk = (wave >> 2) * 64;  // wave's 64-col slab

    const int half = lane >> 4;           // ISA frag layout: K-half select
    const int l16  = lane & 15;           // ISA frag layout: M (A) / N (B)

    v8f acc[2][4] = {};

    for (int k0 = 0; k0 < K; k0 += KT) {
        // ---- A tile (128x32 bf16 = 8KB): async raw copy global -> LDS.
        //      512 b128 chunks, 2 per thread. K-chunk is 32-aligned so the
        //      whole chunk comes from exactly one of the concat sources. ----
#pragma unroll
        for (int i = 0; i < 2; ++i) {
            const int c   = tid + i * 256;
            const int r   = c >> 2;
            const int col = (c & 3) * 8;
            const int gc  = k0 + col;
            const bf16* src = (gc < K0) ? (A0 + (size_t)(tileM + r) * K0 + gc)
                                        : (A1 + (size_t)(tileM + r) * K1 + (gc - K0));
            const unsigned ldsOff = (unsigned)(uintptr_t)&As[r][col];
            asm volatile("global_load_async_to_lds_b128 %0, %1, off"
                         :: "v"(ldsOff),
                            "v"((unsigned long long)(uintptr_t)src)
                         : "memory");
        }
        // ---- B tile (128x32): W rows (output features), fp32 -> bf16 ----
#pragma unroll
        for (int i = 0; i < 2; ++i) {
            const int c   = tid + i * 256;
            const int r   = c >> 2;
            const int col = (c & 3) * 8;
            const float* src = W + (size_t)(tileN + r) * K + k0 + col;
            v8bf tmp;
#pragma unroll
            for (int j = 0; j < 8; ++j) tmp[j] = (bf16)src[j];
            *(v8bf*)&Bs[r][col] = tmp;
            if (k0 + KT < K)
                __builtin_prefetch(src + KT, 0, 1);   // global_prefetch_b8
        }
        asm volatile("s_wait_asynccnt 0x0" ::: "memory");
        __syncthreads();

        // ---- fragments (ISA 7.12.2 layouts) + 8 WMMAs ----
        union frag { v16bf v; v8bf h[2]; };
        frag a[2], b[4];
#pragma unroll
        for (int i = 0; i < 2; ++i) {
            a[i].h[0] = *(const v8bf*)&As[rowblk + i * 16 + l16][half * 8];
            a[i].h[1] = *(const v8bf*)&As[rowblk + i * 16 + l16][half * 8 + 16];
        }
#pragma unroll
        for (int j = 0; j < 4; ++j) {
            b[j].h[0] = *(const v8bf*)&Bs[colblk + j * 16 + l16][half * 8];
            b[j].h[1] = *(const v8bf*)&Bs[colblk + j * 16 + l16][half * 8 + 16];
        }
#pragma unroll
        for (int i = 0; i < 2; ++i)
#pragma unroll
            for (int j = 0; j < 4; ++j)
                acc[i][j] = __builtin_amdgcn_wmma_f32_16x16x32_bf16(
                    false, a[i].v, false, b[j].v, (short)0, acc[i][j], false, false);
        __syncthreads();
    }

    // ---- epilogue: D layout VGPR r -> row +r+8*half, col l16 ----
#pragma unroll
    for (int i = 0; i < 2; ++i) {
#pragma unroll
        for (int j = 0; j < 4; ++j) {
            const int gn = tileN + colblk + j * 16 + l16;
            const float bi = bias ? bias[gn] : 0.f;
#pragma unroll
            for (int r = 0; r < 8; ++r) {
                const int gm = tileM + rowblk + i * 16 + r + 8 * half;
                float v = acc[i][j][r] + bi;
                if (relu) v = fmaxf(v, 0.f);
                if (outF) outF[(size_t)gm * N + gn] = v;
                if (outH) outH[(size_t)gm * N + gn] = (bf16)v;
            }
        }
    }
}

// ---------------------------------------------------------------------------
// prob kernel: per batch row b,
//   p = sum_j softmax_j(logits) * sigmoid(pw_j * (inp_j - mean_j))
//   probab[left]  = pparent * (1 - p);  probab[right] = pparent * p
// ---------------------------------------------------------------------------
__global__ __launch_bounds__(256)
void k_prob(const float* __restrict__ logits, int N,
            const bf16* __restrict__ A0, int K0,
            const bf16* __restrict__ A1, int K1,
            const float* __restrict__ pw,
            const float* __restrict__ mean0,
            const float* __restrict__ mean1,
            const float* __restrict__ pparent,
            float* __restrict__ pleft,
            float* __restrict__ pright)
{
    const int b = blockIdx.x;
    const int t = threadIdx.x;
    const float* lrow = logits + (size_t)b * N;

    __shared__ float redA[256];
    __shared__ float redB[256];

    float mx = -3.0e38f;
    for (int j = t; j < N; j += 256) mx = fmaxf(mx, lrow[j]);
    redA[t] = mx;
    __syncthreads();
    for (int w = 128; w > 0; w >>= 1) {
        if (t < w) redA[t] = fmaxf(redA[t], redA[t + w]);
        __syncthreads();
    }
    mx = redA[0];
    __syncthreads();

    float se = 0.f, ws = 0.f;
    for (int j = t; j < N; j += 256) {
        const float e   = __expf(lrow[j] - mx);
        const float inp = (j < K0) ? (float)A0[(size_t)b * K0 + j]
                                   : (float)A1[(size_t)b * K1 + (j - K0)];
        const float m   = (j < K0) ? mean0[j] : mean1[j - K0];
        const float d   = 1.f / (1.f + __expf(-(pw[j] * (inp - m))));
        se += e;
        ws += e * d;
    }
    redA[t] = se;
    redB[t] = ws;
    __syncthreads();
    for (int w = 128; w > 0; w >>= 1) {
        if (t < w) { redA[t] += redA[t + w]; redB[t] += redB[t + w]; }
        __syncthreads();
    }
    if (t == 0) {
        const float p  = redB[0] / redA[0];
        const float pp = pparent ? pparent[b] : 1.0f;
        pleft[b]  = pp * (1.f - p);
        pright[b] = pp * p;
    }
}

// ---------------------------------------------------------------------------
// final: actions = leaf_p @ leaf_out, then softmax over 64 outputs
// ---------------------------------------------------------------------------
__global__ __launch_bounds__(64)
void k_final(const float* __restrict__ probab,
             const float* __restrict__ leaf_out,
             float* __restrict__ out)
{
    const int b = blockIdx.x;
    const int o = threadIdx.x;   // 0..63
    float a = 0.f;
#pragma unroll
    for (int i = 0; i < NLEAF; ++i)
        a += probab[(size_t)(15 + i) * BATCH + b] * leaf_out[i * OUTD + o];

    __shared__ float red[64];
    red[o] = a;
    __syncthreads();
    for (int w = 32; w > 0; w >>= 1) {
        if (o < w) red[o] = fmaxf(red[o], red[o + w]);
        __syncthreads();
    }
    const float mx = red[0];
    __syncthreads();
    const float e = __expf(a - mx);
    red[o] = e;
    __syncthreads();
    for (int w = 32; w > 0; w >>= 1) {
        if (o < w) red[o] += red[o + w];
        __syncthreads();
    }
    out[(size_t)b * OUTD + o] = e / red[0];
}

// ---------------------------------------------------------------------------
// host orchestration
// ---------------------------------------------------------------------------
extern "C" void kernel_launch(void* const* d_in, const int* in_sizes, int n_in,
                              void* d_out, int out_size, void* d_ws, size_t ws_size,
                              hipStream_t stream)
{
    const float* x      = (const float*)d_in[0];   // [B, 512]
    const float* calcW0 = (const float*)d_in[1];   // [512, 512]
    const float* calcb0 = (const float*)d_in[2];   // [512]
    const float* pw0    = (const float*)d_in[3];   // [512]
    const float* attnW0 = (const float*)d_in[4];   // [512, 512]
    const float* attnb0 = (const float*)d_in[5];   // [512]
    const float* calcW  = (const float*)d_in[6];   // [14, 512, 1024]
    const float* calcb  = (const float*)d_in[7];   // [14, 512]
    const float* pw     = (const float*)d_in[8];   // [14, 1024]
    const float* attnW  = (const float*)d_in[9];   // [14, 1024, 1024]
    const float* attnb  = (const float*)d_in[10];  // [14, 1024]
    const float* leafO  = (const float*)d_in[11];  // [16, 64]
    float* out = (float*)d_out;                    // [B, 64]

    // workspace carve-up (~101 MB)
    char* w = (char*)d_ws;
    bf16* xh = (bf16*)w;              w += (size_t)BATCH * DIM * 2;          // x in bf16
    bf16* outsH = (bf16*)w;           w += (size_t)7 * BATCH * DIM * 2;      // outs[0..6] bf16
    float* logits = (float*)w;        w += (size_t)BATCH * 1024 * 4;         // attn logits (reused)
    float* xmean = (float*)w;         w += (size_t)DIM * 4;                  // column means of x
    float* omean = (float*)w;         w += (size_t)7 * DIM * 4;              // column means of outs[0..6]
    float* probab = (float*)w;        w += (size_t)31 * BATCH * 4;           // probab[1..30]

    const dim3 blk256(256);

    k_f32_to_bf16<<<dim3((BATCH * DIM + 255) / 256), blk256, 0, stream>>>(
        x, xh, (unsigned)(BATCH * DIM));
    k_colmean_bf16<<<dim3(DIM), blk256, 0, stream>>>(xh, BATCH, DIM, xmean);

    // --- root node (K = 512) ---
    k_gemm_bf16<<<dim3(BATCH / MT, DIM / NT), blk256, 0, stream>>>(
        xh, DIM, (const bf16*)nullptr, 0, calcW0, calcb0,
        (float*)nullptr, outsH /*outs[0]*/, DIM, 1);
    k_colmean_bf16<<<dim3(DIM), blk256, 0, stream>>>(outsH, BATCH, DIM, omean);
    k_gemm_bf16<<<dim3(BATCH / MT, DIM / NT), blk256, 0, stream>>>(
        xh, DIM, (const bf16*)nullptr, 0, attnW0, attnb0,
        logits, (bf16*)nullptr, DIM, 0);
    k_prob<<<dim3(BATCH), blk256, 0, stream>>>(
        logits, DIM, xh, DIM, (const bf16*)nullptr, 0,
        pw0, xmean, (const float*)nullptr, (const float*)nullptr,
        probab + (size_t)1 * BATCH, probab + (size_t)2 * BATCH);

    // --- internal nodes 1..14 (K = 1024). outs only needed for nodes 1..6. ---
    for (int n = 1; n <= 14; ++n) {
        const int parent = (n - 1) / 2;
        const bf16* a0 = outsH + (size_t)parent * BATCH * DIM;   // outs[parent]
        const bf16* a1 = xh;                                     // x

        if (n <= 6) {
            k_gemm_bf16<<<dim3(BATCH / MT, DIM / NT), blk256, 0, stream>>>(
                a0, DIM, a1, DIM,
                calcW + (size_t)(n - 1) * DIM * 1024,
                calcb + (size_t)(n - 1) * DIM,
                (float*)nullptr, outsH + (size_t)n * BATCH * DIM, DIM, 1);
            k_colmean_bf16<<<dim3(DIM), blk256, 0, stream>>>(
                outsH + (size_t)n * BATCH * DIM, BATCH, DIM, omean + (size_t)n * DIM);
        }

        k_gemm_bf16<<<dim3(BATCH / MT, 1024 / NT), blk256, 0, stream>>>(
            a0, DIM, a1, DIM,
            attnW + (size_t)(n - 1) * 1024 * 1024,
            attnb + (size_t)(n - 1) * 1024,
            logits, (bf16*)nullptr, 1024, 0);

        k_prob<<<dim3(BATCH), blk256, 0, stream>>>(
            logits, 1024, a0, DIM, a1, DIM,
            pw + (size_t)(n - 1) * 1024,
            omean + (size_t)parent * DIM, xmean,
            probab + (size_t)n * BATCH,
            probab + (size_t)(2 * n + 1) * BATCH,
            probab + (size_t)(2 * n + 2) * BATCH);
    }

    k_final<<<dim3(BATCH), dim3(64), 0, stream>>>(probab, leafO, out);
}
